// DecoderBlock_21569325760830
// MI455X (gfx1250) — matrix operand order repaired and checked by reading, not compiled, not run on hardware
//
#include <hip/hip_runtime.h>
#include <hip/hip_bf16.h>

// ---------------- problem constants ----------------
#define B_  2
#define S_  2048
#define D_  1024
#define H_  16
#define HD_ 64
#define DF_ 4096
#define BS_ (B_*S_)

typedef __bf16 bf16;
typedef __attribute__((ext_vector_type(16))) __bf16 bf16x16;
typedef __attribute__((ext_vector_type(8)))  float  f32x8;

union FragU { bf16x16 v; unsigned int u[8]; };

// =====================================================================
// f32 -> bf16 convert (weights / encoder states)
// =====================================================================
__global__ void cvt_f32_bf16(const float* __restrict__ in, bf16* __restrict__ out, int n) {
    int i = blockIdx.x * blockDim.x + threadIdx.x;
    int stride = gridDim.x * blockDim.x;
    for (; i < n; i += stride) out[i] = (bf16)in[i];
}

// =====================================================================
// RMSNorm: one block per row of D=1024, write bf16
// =====================================================================
__global__ __launch_bounds__(256)
void rmsnorm_kernel(const float* __restrict__ x, const float* __restrict__ w,
                    bf16* __restrict__ out) {
    __shared__ float red[256];
    const int row = blockIdx.x;
    const float* xr = x + (size_t)row * D_;
    float ss = 0.f;
    for (int i = threadIdx.x; i < D_; i += 256) { float v = xr[i]; ss += v * v; }
    red[threadIdx.x] = ss;
    __syncthreads();
    for (int o = 128; o > 0; o >>= 1) {
        if (threadIdx.x < o) red[threadIdx.x] += red[threadIdx.x + o];
        __syncthreads();
    }
    const float rms = rsqrtf(red[0] / (float)D_ + 1e-6f);
    bf16* orow = out + (size_t)row * D_;
    for (int i = threadIdx.x; i < D_; i += 256) orow[i] = (bf16)(xr[i] * rms * w[i]);
}

// =====================================================================
// RoPE in-place on bf16 [B,S,H*HD]; freqs are (S, HD/2)
// =====================================================================
__global__ __launch_bounds__(256)
void rope_kernel(bf16* __restrict__ q, const float* __restrict__ fc,
                 const float* __restrict__ fs) {
    int idx = blockIdx.x * blockDim.x + threadIdx.x;   // B*S*H*32 threads
    int i = idx & 31;
    int h = (idx >> 5) & (H_ - 1);
    int s = (idx >> 9) & (S_ - 1);
    int b = idx >> 20;
    float c = fc[s * 32 + i], sn = fs[s * 32 + i];
    size_t base = ((size_t)(b * S_ + s)) * D_ + h * HD_;
    float x1 = (float)q[base + i];
    float x2 = (float)q[base + 32 + i];
    q[base + i]      = (bf16)(x1 * c - x2 * sn);
    q[base + 32 + i] = (bf16)(x1 * sn + x2 * c);
}

// =====================================================================
// Tiled bf16 GEMM via v_wmma_f32_16x16x32_bf16
//   block tile 128x128, 8 waves (4x2), wave tile 32x64 (2x4 wmma tiles)
//   A [MxK] row-major bf16, B [KxN] row-major bf16
// EPI 0: outBf = A@B   (bf16)
// EPI 1: outF  = resid + ls[col]*(A@B)    (fp32, in-place ok)
// EPI 2: outBf = silu(A@B0) * (A@B1)      (bf16, SwiGLU fused)
// =====================================================================
#define BM 128
#define BN 128
#define BK 32
#define BTS 34   // transposed-B LDS stride (keeps 4B alignment for b32 reads)

template<int EPI>
__global__ __launch_bounds__(256)
void gemm_bf16(const bf16* __restrict__ A, const bf16* __restrict__ Bm0,
               const bf16* __restrict__ Bm1, int M, int N, int K,
               bf16* __restrict__ outBf, float* __restrict__ outF,
               const float* __restrict__ resid, const float* __restrict__ ls) {
    __shared__ unsigned short ldsA[BM * BK];                    // [row][k]
    __shared__ unsigned short ldsB0[BN * BTS];                  // [n][k] (transposed)
    __shared__ unsigned short ldsB1[(EPI == 2) ? (BN * BTS) : 2];

    const int tid  = threadIdx.x;
    const int lane = tid & 31;
    const int wv   = tid >> 5;        // 0..7
    const int wm   = wv >> 1;         // 0..3  (M direction)
    const int wn   = wv & 1;          // 0..1  (N direction)
    const int g    = lane >> 4;       // half-wave group
    const int ln   = lane & 15;

    const int bm = blockIdx.y * BM;
    const int bn = blockIdx.x * BN;

    f32x8 acc0[2][4];
    f32x8 acc1[2][4];
#pragma unroll
    for (int mt = 0; mt < 2; ++mt)
#pragma unroll
        for (int nt = 0; nt < 4; ++nt) { acc0[mt][nt] = {}; acc1[mt][nt] = {}; }

    // staging assignments (256 threads)
    const int aRow = tid >> 1;            // 0..127
    const int aCol = (tid & 1) * 16;      // 0 / 16
    const int bK   = tid >> 3;            // 0..31
    const int bCol = (tid & 7) * 16;      // 0..112

    for (int k0 = 0; k0 < K; k0 += BK) {
        // ---- stage A tile (row-major in K) ----
        const bf16* aptr = A + (size_t)(bm + aRow) * K + k0 + aCol;
        {
            uint4 a0 = ((const uint4*)aptr)[0];
            uint4 a1 = ((const uint4*)aptr)[1];
            ((uint4*)&ldsA[aRow * BK + aCol])[0] = a0;
            ((uint4*)&ldsA[aRow * BK + aCol])[1] = a1;
        }
        // ---- stage B tile transposed: lds[n][k] ----
        const bf16* bptr = Bm0 + (size_t)(k0 + bK) * N + bn + bCol;
        {
            uint4 b0 = ((const uint4*)bptr)[0];
            uint4 b1 = ((const uint4*)bptr)[1];
            const unsigned short* bs = (const unsigned short*)&b0;
#pragma unroll
            for (int i = 0; i < 8; ++i) ldsB0[(bCol + i) * BTS + bK] = bs[i];
            bs = (const unsigned short*)&b1;
#pragma unroll
            for (int i = 0; i < 8; ++i) ldsB0[(bCol + 8 + i) * BTS + bK] = bs[i];
        }
        if (EPI == 2) {
            const bf16* b1p = Bm1 + (size_t)(k0 + bK) * N + bn + bCol;
            uint4 b0 = ((const uint4*)b1p)[0];
            uint4 b1 = ((const uint4*)b1p)[1];
            const unsigned short* bs = (const unsigned short*)&b0;
#pragma unroll
            for (int i = 0; i < 8; ++i) ldsB1[(bCol + i) * BTS + bK] = bs[i];
            bs = (const unsigned short*)&b1;
#pragma unroll
            for (int i = 0; i < 8; ++i) ldsB1[(bCol + 8 + i) * BTS + bK] = bs[i];
        }
        if (k0 + BK < K) {                       // hint next chunk into cache
            __builtin_prefetch(aptr + BK, 0, 0);               // global_prefetch_b8
            __builtin_prefetch(bptr + (size_t)BK * N, 0, 0);
        }
        __syncthreads();

        // ---- A fragments (16x32 bf16): lane = M row, quarter K pattern ----
        FragU af[2];
#pragma unroll
        for (int mt = 0; mt < 2; ++mt) {
            const int row = wm * 32 + mt * 16 + ln;
#pragma unroll
            for (int j = 0; j < 8; ++j) {
                const int kk = ((j < 4) ? 0 : 16) + g * 8 + (j & 3) * 2;
                af[mt].u[j] = *(const unsigned int*)&ldsA[row * BK + kk];
            }
        }
        // ---- B fragments (32x16 bf16): lane = N col, quarter K pattern ----
        FragU bf0[4], bf1[4];
#pragma unroll
        for (int nt = 0; nt < 4; ++nt) {
            const int col = wn * 64 + nt * 16 + ln;
#pragma unroll
            for (int j = 0; j < 8; ++j) {
                const int kk = ((j < 4) ? 0 : 16) + g * 8 + (j & 3) * 2;
                bf0[nt].u[j] = *(const unsigned int*)&ldsB0[col * BTS + kk];
                if (EPI == 2) bf1[nt].u[j] = *(const unsigned int*)&ldsB1[col * BTS + kk];
            }
        }
#pragma unroll
        for (int mt = 0; mt < 2; ++mt)
#pragma unroll
            for (int nt = 0; nt < 4; ++nt) {
                acc0[mt][nt] = __builtin_amdgcn_wmma_f32_16x16x32_bf16(
                    false, af[mt].v, false, bf0[nt].v, (short)0, acc0[mt][nt], false, false);
                if (EPI == 2)
                    acc1[mt][nt] = __builtin_amdgcn_wmma_f32_16x16x32_bf16(
                        false, af[mt].v, false, bf1[nt].v, (short)0, acc1[mt][nt], false, false);
            }
        __syncthreads();
    }

    // ---- epilogue: C layout VGPR r -> M = r + 8*g, N = ln ----
#pragma unroll
    for (int mt = 0; mt < 2; ++mt)
#pragma unroll
        for (int nt = 0; nt < 4; ++nt)
#pragma unroll
            for (int r = 0; r < 8; ++r) {
                const int row = bm + wm * 32 + mt * 16 + r + 8 * g;
                const int col = bn + wn * 64 + nt * 16 + ln;
                const size_t idx = (size_t)row * N + col;
                const float v = acc0[mt][nt][r];
                if (EPI == 0) {
                    outBf[idx] = (bf16)v;
                } else if (EPI == 1) {
                    outF[idx] = resid[idx] + ls[col] * v;
                } else {
                    const float v3 = acc1[mt][nt][r];
                    const float si = v / (1.f + __expf(-v));
                    outBf[idx] = (bf16)(si * v3);
                }
            }
}

// =====================================================================
// Flash attention: 1 wave per (b, h, 16-query tile); 32-key chunks.
// QK^T and PV both via v_wmma_f32_16x16x32_bf16.
// =====================================================================
template<bool CAUSAL>
__global__ __launch_bounds__(32)
void attn_kernel(const bf16* __restrict__ Q, const bf16* __restrict__ Kb,
                 const bf16* __restrict__ Vb, bf16* __restrict__ O) {
    __shared__ unsigned short ldsK[32 * 64];    // [key][dim]
    __shared__ unsigned short ldsVt[64 * 34];   // [dim][key]  (transposed V)
    __shared__ unsigned short ldsP[16 * 32];    // [row][key]  (probs bounce)

    const int lane = threadIdx.x;
    const int g = lane >> 4, ln = lane & 15;
    const int qt = blockIdx.x, h = blockIdx.y, b = blockIdx.z;
    const int qbase = qt * 16;

    // Q fragments: two 16x32 chunks covering HD=64
    FragU qf[2];
    {
        const bf16* qrow = Q + ((size_t)(b * S_ + qbase + ln)) * D_ + h * HD_;
#pragma unroll
        for (int c = 0; c < 2; ++c)
#pragma unroll
            for (int j = 0; j < 8; ++j) {
                const int kk = c * 32 + ((j < 4) ? 0 : 16) + g * 8 + (j & 3) * 2;
                qf[c].u[j] = *(const unsigned int*)(qrow + kk);
            }
    }

    f32x8 oacc[4];
#pragma unroll
    for (int t = 0; t < 4; ++t) oacc[t] = {};
    float rmax[8], rsum[8];
#pragma unroll
    for (int r = 0; r < 8; ++r) { rmax[r] = -3.0e38f; rsum[r] = 0.f; }

    const int limit = CAUSAL ? (qbase + 16) : S_;
    for (int kb = 0; kb < limit; kb += 32) {
        // ---- stage K (row-major) and V (transposed) into LDS ----
        {
            const size_t krow = ((size_t)(b * S_ + kb + lane)) * D_ + h * HD_;
            const uint4* ksrc = (const uint4*)(Kb + krow);
            uint4* kdst = (uint4*)&ldsK[lane * 64];
#pragma unroll
            for (int c = 0; c < 8; ++c) kdst[c] = ksrc[c];
            const uint4* vsrc = (const uint4*)(Vb + krow);
#pragma unroll
            for (int c = 0; c < 8; ++c) {
                uint4 vv = vsrc[c];
                const unsigned short* vs = (const unsigned short*)&vv;
#pragma unroll
                for (int e = 0; e < 8; ++e)
                    ldsVt[(c * 8 + e) * 34 + lane] = vs[e];
            }
        }
        asm volatile("s_wait_dscnt 0" ::: "memory");

        // ---- scores: two 16x16 tiles (keys [0,16) and [16,32) of chunk) ----
        f32x8 sc[2];
#pragma unroll
        for (int t = 0; t < 2; ++t) {
            FragU bk0, bk1;
            const int key = t * 16 + ln;
#pragma unroll
            for (int j = 0; j < 8; ++j) {
                const int kd = ((j < 4) ? 0 : 16) + g * 8 + (j & 3) * 2;
                bk0.u[j] = *(const unsigned int*)&ldsK[key * 64 + kd];
                bk1.u[j] = *(const unsigned int*)&ldsK[key * 64 + 32 + kd];
            }
            f32x8 z = {};
            f32x8 a = __builtin_amdgcn_wmma_f32_16x16x32_bf16(
                false, qf[0].v, false, bk0.v, (short)0, z, false, false);
            a = __builtin_amdgcn_wmma_f32_16x16x32_bf16(
                false, qf[1].v, false, bk1.v, (short)0, a, false, false);
            sc[t] = a;
        }

        // ---- online softmax (rows live in lane-groups of 16) ----
#pragma unroll
        for (int r = 0; r < 8; ++r) {
            float v0 = sc[0][r] * 0.125f;
            float v1 = sc[1][r] * 0.125f;
            if (CAUSAL) {
                const int qrow = qbase + r + 8 * g;
                if (kb + ln > qrow)      v0 = -3.0e38f;
                if (kb + 16 + ln > qrow) v1 = -3.0e38f;
            }
            float m = fmaxf(v0, v1);
#pragma unroll
            for (int off = 8; off; off >>= 1) m = fmaxf(m, __shfl_xor(m, off, 16));
            const float nm = fmaxf(rmax[r], m);
            const float corr = __expf(rmax[r] - nm);
            rmax[r] = nm;
            const float p0 = __expf(v0 - nm);
            const float p1 = __expf(v1 - nm);
            float rs = p0 + p1;
#pragma unroll
            for (int off = 8; off; off >>= 1) rs += __shfl_xor(rs, off, 16);
            rsum[r] = rsum[r] * corr + rs;
#pragma unroll
            for (int t = 0; t < 4; ++t) oacc[t][r] *= corr;
            const int prow = r + 8 * g;
            bf16 pb0 = (bf16)p0, pb1 = (bf16)p1;
            ldsP[prow * 32 + ln]      = *(unsigned short*)&pb0;
            ldsP[prow * 32 + 16 + ln] = *(unsigned short*)&pb1;
        }
        asm volatile("s_wait_dscnt 0" ::: "memory");

        // ---- PV: P (16x32) as A-fragment, V chunks as B ----
        FragU pf;
#pragma unroll
        for (int j = 0; j < 8; ++j) {
            const int kk = ((j < 4) ? 0 : 16) + g * 8 + (j & 3) * 2;
            pf.u[j] = *(const unsigned int*)&ldsP[ln * 32 + kk];
        }
#pragma unroll
        for (int t = 0; t < 4; ++t) {
            FragU vf;
#pragma unroll
            for (int j = 0; j < 8; ++j) {
                const int kd = ((j < 4) ? 0 : 16) + g * 8 + (j & 3) * 2;   // key index in chunk
                vf.u[j] = *(const unsigned int*)&ldsVt[(t * 16 + ln) * 34 + kd];
            }
            oacc[t] = __builtin_amdgcn_wmma_f32_16x16x32_bf16(
                false, pf.v, false, vf.v, (short)0, oacc[t], false, false);
        }
        asm volatile("s_wait_dscnt 0" ::: "memory");
    }

    // ---- normalize and store ----
#pragma unroll
    for (int r = 0; r < 8; ++r) rsum[r] = (rsum[r] > 0.f) ? 1.f / rsum[r] : 0.f;
#pragma unroll
    for (int t = 0; t < 4; ++t)
#pragma unroll
        for (int r = 0; r < 8; ++r) {
            const int qrow = qbase + r + 8 * g;
            const size_t idx = ((size_t)(b * S_ + qrow)) * D_ + h * HD_ + t * 16 + ln;
            O[idx] = (bf16)(oacc[t][r] * rsum[r]);
        }
}

// =====================================================================
// Orchestration
// =====================================================================
extern "C" void kernel_launch(void* const* d_in, const int* in_sizes, int n_in,
                              void* d_out, int out_size, void* d_ws, size_t ws_size,
                              hipStream_t stream) {
    const float* x    = (const float*)d_in[0];
    const float* enc  = (const float*)d_in[1];
    const float* fcos = (const float*)d_in[2];
    const float* fsin = (const float*)d_in[3];
    const float* n1w  = (const float*)d_in[4];
    const float* n2w  = (const float*)d_in[5];
    const float* n3w  = (const float*)d_in[6];
    const float* ls1  = (const float*)d_in[7];
    const float* ls2  = (const float*)d_in[8];
    const float* ls3  = (const float*)d_in[9];
    const float* wq_s = (const float*)d_in[10];
    const float* wk_s = (const float*)d_in[11];
    const float* wv_s = (const float*)d_in[12];
    const float* wo_s = (const float*)d_in[13];
    const float* wq_c = (const float*)d_in[14];
    const float* wk_c = (const float*)d_in[15];
    const float* wv_c = (const float*)d_in[16];
    const float* wv_c_unused = wv_c; (void)wv_c_unused;
    const float* wo_c = (const float*)d_in[17];
    const float* w1   = (const float*)d_in[18];
    const float* w3   = (const float*)d_in[19];
    const float* w2   = (const float*)d_in[20];
    float* xo = (float*)d_out;          // running residual, fp32

    char* ws = (char*)d_ws;
    const size_t SZ = (size_t)BS_ * D_ * 2;   // 8 MB unit
    bf16* hbf  = (bf16*)(ws + 0 * SZ);
    bf16* qbf  = (bf16*)(ws + 1 * SZ);
    bf16* kbf  = (bf16*)(ws + 2 * SZ);
    bf16* vbf  = (bf16*)(ws + 3 * SZ);
    bf16* obf  = (bf16*)(ws + 4 * SZ);
    bf16* wA   = (bf16*)(ws + 5 * SZ);        // D*DF bf16 scratch (8 MB)
    bf16* wB   = (bf16*)(ws + 6 * SZ);
    bf16* gate = (bf16*)(ws + 7 * SZ);        // BS*DF bf16 (32 MB); also ebf
    bf16* ebf  = gate;

    hipMemcpyAsync(xo, x, (size_t)BS_ * D_ * sizeof(float),
                   hipMemcpyDeviceToDevice, stream);

    const dim3 gridD(D_ / BN, BS_ / BM);      // (8, 32)
    const dim3 gridDF(DF_ / BN, BS_ / BM);    // (32, 32)
    const dim3 gridAtt(S_ / 16, H_, B_);
    const int ropeBlocks = (B_ * S_ * H_ * 32) / 256;

#define CVT(src, dst, n) cvt_f32_bf16<<<1024, 256, 0, stream>>>((src), (dst), (n))

    // ---------- self attention ----------
    rmsnorm_kernel<<<BS_, 256, 0, stream>>>(xo, n1w, hbf);
    CVT(wq_s, wA, D_ * D_);
    gemm_bf16<0><<<gridD, 256, 0, stream>>>(hbf, wA, nullptr, BS_, D_, D_, qbf, nullptr, nullptr, nullptr);
    CVT(wk_s, wA, D_ * D_);
    gemm_bf16<0><<<gridD, 256, 0, stream>>>(hbf, wA, nullptr, BS_, D_, D_, kbf, nullptr, nullptr, nullptr);
    CVT(wv_s, wA, D_ * D_);
    gemm_bf16<0><<<gridD, 256, 0, stream>>>(hbf, wA, nullptr, BS_, D_, D_, vbf, nullptr, nullptr, nullptr);
    rope_kernel<<<ropeBlocks, 256, 0, stream>>>(qbf, fcos, fsin);
    rope_kernel<<<ropeBlocks, 256, 0, stream>>>(kbf, fcos, fsin);
    attn_kernel<true><<<gridAtt, 32, 0, stream>>>(qbf, kbf, vbf, obf);
    CVT(wo_s, wA, D_ * D_);
    gemm_bf16<1><<<gridD, 256, 0, stream>>>(obf, wA, nullptr, BS_, D_, D_, nullptr, xo, xo, ls1);

    // ---------- cross attention ----------
    rmsnorm_kernel<<<BS_, 256, 0, stream>>>(xo, n2w, hbf);
    CVT(enc, ebf, BS_ * D_);
    CVT(wq_c, wA, D_ * D_);
    gemm_bf16<0><<<gridD, 256, 0, stream>>>(hbf, wA, nullptr, BS_, D_, D_, qbf, nullptr, nullptr, nullptr);
    CVT(wk_c, wA, D_ * D_);
    gemm_bf16<0><<<gridD, 256, 0, stream>>>(ebf, wA, nullptr, BS_, D_, D_, kbf, nullptr, nullptr, nullptr);
    CVT(wv_c, wA, D_ * D_);
    gemm_bf16<0><<<gridD, 256, 0, stream>>>(ebf, wA, nullptr, BS_, D_, D_, vbf, nullptr, nullptr, nullptr);
    attn_kernel<false><<<gridAtt, 32, 0, stream>>>(qbf, kbf, vbf, obf);
    CVT(wo_c, wA, D_ * D_);
    gemm_bf16<1><<<gridD, 256, 0, stream>>>(obf, wA, nullptr, BS_, D_, D_, nullptr, xo, xo, ls2);

    // ---------- FFN (SwiGLU) ----------
    rmsnorm_kernel<<<BS_, 256, 0, stream>>>(xo, n3w, hbf);
    CVT(w1, wA, D_ * DF_);
    CVT(w3, wB, D_ * DF_);
    gemm_bf16<2><<<gridDF, 256, 0, stream>>>(hbf, wA, wB, BS_, DF_, D_, gate, nullptr, nullptr, nullptr);
    CVT(w2, wA, DF_ * D_);
    gemm_bf16<1><<<gridD, 256, 0, stream>>>(gate, wA, nullptr, BS_, D_, DF_, nullptr, xo, xo, ls3);
#undef CVT
}
